// Dot_attention_5076651344174
// MI455X (gfx1250) — compile-verified, matrix operand order checked by reading, not run on hardware
//
#include <hip/hip_runtime.h>

// ---------------------------------------------------------------------------
// Multi-head dot-product attention for MI455X (gfx1250, wave32, WMMA).
// out  = softmax(mask(QWq (KWk)^T / sqrt(dh))) @ (VWv)   per head
// Pipeline:
//   1) proj_kernel : q/k/v [8192,1024] x W^T -> qp,kp (f16, [BH,S,dh]) and
//                    vpT (f16, [BH,dh,S])   -- f32->f16 convert on load
//   2) attn_kernel : scores (WMMA f16) + scale + valid_len mask + softmax,
//                    score band kept in registers, writes attn f32 (output)
//   3) out_kernel  : attn (f32->f16 on load) @ vpT -> out f32
// ---------------------------------------------------------------------------

typedef __attribute__((ext_vector_type(16))) _Float16 v16h;
typedef __attribute__((ext_vector_type(8)))  _Float16 v8h;
typedef __attribute__((ext_vector_type(8)))  float    v8f;
typedef __attribute__((ext_vector_type(4)))  float    v4f;

#define NB   8
#define NS   1024
#define ND   1024
#define NH   16
#define NDH  64
#define NBH  (NB * NH)   // 128

static __device__ __forceinline__ v8f wmma16(v16h a, v16h b, v8f c) {
  // (neg_a, A, neg_b, B, c_mod, C, reuse_a, reuse_b)
  return __builtin_amdgcn_wmma_f32_16x16x32_f16(false, a, false, b, (short)0, c,
                                                false, false);
}

// A/B fragment per ISA 7.12.2: for lane half h = lane/16, v16h elements
// 0..7  = K = base+8h .. base+8h+7   (contiguous)
// 8..15 = K = base+16+8h .. +7       (contiguous)
// p0/p1 point at those two runs.
static __device__ __forceinline__ v16h frag_f32(const float* __restrict__ p0,
                                                const float* __restrict__ p1) {
  v4f a0 = *(const v4f*)p0;
  v4f a1 = *(const v4f*)(p0 + 4);
  v4f b0 = *(const v4f*)p1;
  v4f b1 = *(const v4f*)(p1 + 4);
  v16h r;
#pragma unroll
  for (int i = 0; i < 4; ++i) {
    r[i]      = (_Float16)a0[i];
    r[4 + i]  = (_Float16)a1[i];
    r[8 + i]  = (_Float16)b0[i];
    r[12 + i] = (_Float16)b1[i];
  }
  return r;
}

static __device__ __forceinline__ v16h frag_f16(const _Float16* __restrict__ p0,
                                                const _Float16* __restrict__ p1) {
  v8h a = *(const v8h*)p0;   // global_load_b128
  v8h b = *(const v8h*)p1;   // global_load_b128
  v16h r;
#pragma unroll
  for (int i = 0; i < 8; ++i) { r[i] = a[i]; r[8 + i] = b[i]; }
  return r;
}

// ---------------------------------------------------------------------------
// Kernel 1: projections. Grid (N/64, M/64, 3). 256 thr = 8 waves.
// Wave w: msub = w&3, nsubs = {2*(w>>2), 2*(w>>2)+1}. 64x64 tile per block.
// y[m,n] = dot(X[m,:], W[n,:])  (NT GEMM, both contiguous along K)
// ---------------------------------------------------------------------------
__global__ __launch_bounds__(256) void proj_kernel(
    const float* __restrict__ q, const float* __restrict__ k,
    const float* __restrict__ v, const float* __restrict__ Wq,
    const float* __restrict__ Wk, const float* __restrict__ Wv,
    _Float16* __restrict__ qp, _Float16* __restrict__ kp,
    _Float16* __restrict__ vpT) {
  const int nt = blockIdx.x, mt = blockIdx.y, z = blockIdx.z;
  const float* X = (z == 0) ? q : (z == 1) ? k : v;
  const float* W = (z == 0) ? Wq : (z == 1) ? Wk : Wv;

  const int wave = threadIdx.x >> 5, lane = threadIdx.x & 31;
  const int lh = lane >> 4, lm = lane & 15;
  const int msub = wave & 3, nsub0 = (wave >> 2) * 2;

  const float* arow  = X + (size_t)(mt * 64 + msub * 16 + lm) * ND;
  const float* brow0 = W + (size_t)(nt * 64 + nsub0 * 16 + lm) * ND;
  const float* brow1 = brow0 + (size_t)16 * ND;

  v8f acc0 = {}, acc1 = {};
#pragma unroll 4
  for (int kb = 0; kb < ND; kb += 32) {
    const int o0 = kb + lh * 8, o1 = kb + 16 + lh * 8;
    v16h a  = frag_f32(arow + o0, arow + o1);
    v16h b0 = frag_f32(brow0 + o0, brow0 + o1);
    v16h b1 = frag_f32(brow1 + o0, brow1 + o1);
    acc0 = wmma16(a, b0, acc0);
    acc1 = wmma16(a, b1, acc1);
  }

  // C/D layout: VGPR r, lane L -> M = r + 8*(L/16), N = L%16
#pragma unroll
  for (int r = 0; r < 8; ++r) {
    const int m = mt * 64 + msub * 16 + r + 8 * lh;  // row in [8192]
    const int b = m >> 10, s = m & (NS - 1);
#pragma unroll
    for (int g = 0; g < 2; ++g) {
      const int n = nt * 64 + (nsub0 + g) * 16 + lm;  // col in [1024]
      const int h = n >> 6, j = n & 63;
      const float val = (g == 0) ? acc0[r] : acc1[r];
      const size_t bhix = (size_t)(b * NH + h);
      if (z == 0)
        qp[(bhix * NS + s) * NDH + j] = (_Float16)val;
      else if (z == 1)
        kp[(bhix * NS + s) * NDH + j] = (_Float16)val;
      else
        vpT[(bhix * NDH + j) * NS + s] = (_Float16)val;  // transposed store
    }
  }
}

// ---------------------------------------------------------------------------
// Kernel 2: fused scores + mask + softmax. Grid (S/16, BH). 256 thr = 8 waves.
// Block computes a 16(q)x1024(k) score band; wave w owns cols [128w,128w+128)
// as 8 accumulators, all kept in registers through softmax.
// ---------------------------------------------------------------------------
__global__ __launch_bounds__(256) void attn_kernel(
    const _Float16* __restrict__ qp, const _Float16* __restrict__ kp,
    const int* __restrict__ valid, float* __restrict__ attn) {
  __shared__ float part[16][8];   // per-(row, wave) partials
  __shared__ float rowred[16];    // per-row reduced value

  const int mt = blockIdx.x;      // q band (16 rows)
  const int bh = blockIdx.y;      // batch*head
  const int tid = threadIdx.x;
  const int wave = tid >> 5, lane = tid & 31;
  const int lh = lane >> 4, lm = lane & 15;

  const _Float16* Q  = qp + (size_t)bh * NS * NDH;
  const _Float16* Kp = kp + (size_t)bh * NS * NDH;
  const _Float16* arow = Q + (size_t)(mt * 16 + lm) * NDH;

  v8f acc[8];
#pragma unroll
  for (int i = 0; i < 8; ++i) acc[i] = (v8f){};

#pragma unroll
  for (int kb = 0; kb < NDH; kb += 32) {
    const int o0 = kb + lh * 8, o1 = kb + 16 + lh * 8;
    v16h a = frag_f16(arow + o0, arow + o1);
#pragma unroll
    for (int sub = 0; sub < 8; ++sub) {
      const _Float16* brow =
          Kp + (size_t)(wave * 128 + sub * 16 + lm) * NDH;
      v16h b = frag_f16(brow + o0, brow + o1);
      acc[sub] = wmma16(a, b, acc[sub]);
    }
  }

  // scale by 1/sqrt(64) and mask invalid keys
  const int vlen = valid[bh];
  float x[8][8];  // [sub][r]
#pragma unroll
  for (int sub = 0; sub < 8; ++sub) {
    const int col = wave * 128 + sub * 16 + lm;
    const bool oob = (col >= vlen);
#pragma unroll
    for (int r = 0; r < 8; ++r)
      x[sub][r] = oob ? -1e9f : acc[sub][r] * 0.125f;
  }

  // row max: subs in-register, then 16-lane half via shfl_xor, then waves
  float rloc[8];
#pragma unroll
  for (int r = 0; r < 8; ++r) {
    float m = x[0][r];
#pragma unroll
    for (int sub = 1; sub < 8; ++sub) m = fmaxf(m, x[sub][r]);
#pragma unroll
    for (int d = 1; d < 16; d <<= 1) m = fmaxf(m, __shfl_xor(m, d, 32));
    rloc[r] = m;
  }
  if (lm == 0) {
#pragma unroll
    for (int r = 0; r < 8; ++r) part[r + 8 * lh][wave] = rloc[r];
  }
  __syncthreads();
  if (tid < 16) {
    float m = part[tid][0];
#pragma unroll
    for (int w = 1; w < 8; ++w) m = fmaxf(m, part[tid][w]);
    rowred[tid] = m;
  }
  __syncthreads();
  float rmax[8];
#pragma unroll
  for (int r = 0; r < 8; ++r) rmax[r] = rowred[r + 8 * lh];
  __syncthreads();

  // exp + row sum (same reduction pattern)
#pragma unroll
  for (int r = 0; r < 8; ++r) {
    float s = 0.f;
#pragma unroll
    for (int sub = 0; sub < 8; ++sub) {
      const float e = __expf(x[sub][r] - rmax[r]);
      x[sub][r] = e;
      s += e;
    }
#pragma unroll
    for (int d = 1; d < 16; d <<= 1) s += __shfl_xor(s, d, 32);
    rloc[r] = s;
  }
  if (lm == 0) {
#pragma unroll
    for (int r = 0; r < 8; ++r) part[r + 8 * lh][wave] = rloc[r];
  }
  __syncthreads();
  if (tid < 16) {
    float s = part[tid][0];
#pragma unroll
    for (int w = 1; w < 8; ++w) s += part[tid][w];
    rowred[tid] = 1.0f / s;
  }
  __syncthreads();

  float* dst = attn + ((size_t)bh * NS + (size_t)(mt * 16)) * NS;
#pragma unroll
  for (int r = 0; r < 8; ++r) {
    const int row = r + 8 * lh;
    const float inv = rowred[row];
#pragma unroll
    for (int sub = 0; sub < 8; ++sub) {
      const int col = wave * 128 + sub * 16 + lm;
      dst[(size_t)row * NS + col] = x[sub][r] * inv;
    }
  }
}

// ---------------------------------------------------------------------------
// Kernel 3: out = attn @ vp. Grid (S/64, BH). A = attn f32 (cvt), B = vpT f16.
// N = dh = 64 = full tile width; K = S = 1024.
// ---------------------------------------------------------------------------
__global__ __launch_bounds__(256) void out_kernel(
    const float* __restrict__ attn, const _Float16* __restrict__ vpT,
    float* __restrict__ out) {
  const int mt = blockIdx.x;   // q tile of 64 rows
  const int bh = blockIdx.y;
  const int wave = threadIdx.x >> 5, lane = threadIdx.x & 31;
  const int lh = lane >> 4, lm = lane & 15;
  const int msub = wave & 3, nsub0 = (wave >> 2) * 2;

  const float* A =
      attn + ((size_t)bh * NS + (size_t)(mt * 64 + msub * 16 + lm)) * NS;
  const _Float16* B0 = vpT + ((size_t)bh * NDH + (nsub0 * 16 + lm)) * NS;
  const _Float16* B1 = B0 + (size_t)16 * NS;

  v8f acc0 = {}, acc1 = {};
#pragma unroll 4
  for (int kb = 0; kb < NS; kb += 32) {
    const int o0 = kb + lh * 8, o1 = kb + 16 + lh * 8;
    v16h a  = frag_f32(A + o0, A + o1);
    v16h b0 = frag_f16(B0 + o0, B0 + o1);
    v16h b1 = frag_f16(B1 + o0, B1 + o1);
    acc0 = wmma16(a, b0, acc0);
    acc1 = wmma16(a, b1, acc1);
  }

#pragma unroll
  for (int r = 0; r < 8; ++r) {
    const int m = mt * 64 + msub * 16 + r + 8 * lh;
    float* o = out + ((size_t)bh * NS + m) * NDH;
    o[nsub0 * 16 + lm]      = acc0[r];
    o[nsub0 * 16 + lm + 16] = acc1[r];
  }
}

// ---------------------------------------------------------------------------
extern "C" void kernel_launch(void* const* d_in, const int* in_sizes, int n_in,
                              void* d_out, int out_size, void* d_ws,
                              size_t ws_size, hipStream_t stream) {
  (void)in_sizes; (void)n_in; (void)out_size; (void)ws_size;
  const float* q    = (const float*)d_in[0];
  const float* k    = (const float*)d_in[1];
  const float* v    = (const float*)d_in[2];
  const float* Wq   = (const float*)d_in[3];
  const float* Wk   = (const float*)d_in[4];
  const float* Wv   = (const float*)d_in[5];
  const int*  valid = (const int*)d_in[6];
  // d_in[7] = num_heads (compile-time NH)

  // workspace: qp, kp [BH,S,dh] f16 ; vpT [BH,dh,S] f16  -> 48 MB total
  _Float16* qp  = (_Float16*)d_ws;
  _Float16* kp  = qp + (size_t)NB * NS * ND;
  _Float16* vpT = kp + (size_t)NB * NS * ND;

  float* out  = (float*)d_out;                    // [BH,S,dh]
  float* attn = out + (size_t)NBH * NS * NDH;     // [BH,S,S]

  proj_kernel<<<dim3(ND / 64, NB * NS / 64, 3), 256, 0, stream>>>(
      q, k, v, Wq, Wk, Wv, qp, kp, vpT);
  attn_kernel<<<dim3(NS / 16, NBH), 256, 0, stream>>>(qp, kp, valid, attn);
  out_kernel<<<dim3(NS / 64, NBH), 256, 0, stream>>>(attn, vpT, out);
}
